// CustomDQNNetwork_EdgeAttentionWithNodeEncoding_91156385890412
// MI455X (gfx1250) — matrix operand order repaired and checked by reading, not compiled, not run on hardware
//
#include <hip/hip_runtime.h>
#include <hip/hip_bf16.h>
#include <math.h>

// ---------------------------------------------------------------------------
// Problem constants (from reference): B=32, N=64, D=128, H=2, DH=64
// ---------------------------------------------------------------------------
#define BB 32
#define NN 64
#define DD 128
#define M_EDGE (BB * NN * NN)   // 131072 rows for edge GEMMs
#define M_NODE (BB * NN)        // 2048 rows for node GEMMs

typedef __attribute__((ext_vector_type(16))) _Float16 v16h;
typedef __attribute__((ext_vector_type(8)))  float    v8f;

// ---------------------------------------------------------------------------
// pack_a_kernel: f32 [M,K] row-major -> f16 WMMA A-fragment layout.
// ISA 7.12.2 (16-bit A 16x32): lanes 0-15 row M=lane hold K {0..7,16..23},
// lanes 16-31 row M=lane-16 hold K {8..15,24..31}; element e -> k =
// ((e>>3)<<4) | ((lane>>4)<<3) | (e&7).
// One wave packs one 16x32 fragment; each lane stores its 16 halfs (32B)
// contiguously -> GEMM fragment load is one contiguous 32B/lane access.
// Optional transpose: row m=(b,i,j) reads source row (b,j,i)  (for ec^T).
// ---------------------------------------------------------------------------
__global__ __launch_bounds__(256)
void pack_a_kernel(const float* __restrict__ src, _Float16* __restrict__ dst,
                   int Mrows, int K, int transpose)
{
    int gid  = blockIdx.x * 8 + (threadIdx.x >> 5);   // wave == fragment
    int KT   = K >> 5;
    int nfrag = (Mrows >> 4) * KT;
    if (gid >= nfrag) return;
    int mt = gid / KT;
    int kt = gid - mt * KT;
    int lane = threadIdx.x & 31;

    int row = mt * 16 + (lane & 15);
    if (transpose) {  // row = b*N*N + i*N + j  ->  src row b*N*N + j*N + i
        int b = row >> 12;           // / (N*N)
        int r = row & 4095;
        int i = r >> 6, j = r & 63;
        row = (b << 12) + (j << 6) + i;
    }
    int k0 = kt * 32 + ((lane >> 4) << 3);            // chunks k0..k0+7, +16
    const float* p0 = src + (size_t)row * K + k0;

    v16h o;
#pragma unroll
    for (int e = 0; e < 8; ++e)  o[e]     = (_Float16)p0[e];
#pragma unroll
    for (int e = 0; e < 8; ++e)  o[8 + e] = (_Float16)p0[16 + e];

    *((v16h*)(dst + ((size_t)gid * 32 + lane) * 16)) = o;
}

// ---------------------------------------------------------------------------
// pack_b_kernel: f32 weight -> f16 WMMA B-fragment layout (32x16 per frag).
// B layout: lane 0-15 col N=lane holds K {0..7,16..23}, lanes 16-31
// col N=lane-16 hold K {8..15,24..31} (same k order as A).
//   Bt[k][n] = mode ? w[n*ldw + off + k]   (y = x @ w^T, torch Linear)
//                   : w[(k+off)*ldw + n]   (y = x @ w,   attention proj)
// N fixed at 128 -> 8 n-fragments per kt; frag index = kt*8 + nt.
// ---------------------------------------------------------------------------
__global__ __launch_bounds__(256)
void pack_b_kernel(const float* __restrict__ w, _Float16* __restrict__ dst,
                   int K, int ldw, int off, int mode)
{
    int gid = blockIdx.x * 8 + (threadIdx.x >> 5);
    int KT  = K >> 5;
    if (gid >= KT * 8) return;
    int kt = gid >> 3, nt = gid & 7;
    int lane = threadIdx.x & 31;
    int n = nt * 16 + (lane & 15);
    int kbase = kt * 32 + ((lane >> 4) << 3);

    v16h o;
#pragma unroll
    for (int e = 0; e < 16; ++e) {
        int k = kbase + (e & 7) + ((e >> 3) << 4);
        float v = mode ? w[(size_t)n * ldw + off + k]
                       : w[(size_t)(k + off) * ldw + n];
        o[e] = (_Float16)v;
    }
    *((v16h*)(dst + ((size_t)gid * 32 + lane) * 16)) = o;
}

// ---------------------------------------------------------------------------
// gemm_n128_kernel: C[M,128] = A[M,K] * B[K,128] (+bias +row/col broadcast)
// A,B pre-packed f16 fragments, f32 accumulate via v_wmma_f32_16x16x32_f16.
// Block: 256 threads = 8 wave32; wave w owns rows [blk*128+16w, +16), all
// 128 cols -> 8 v8f accumulators. KSTEPS compile-time -> full unroll.
// The whole B panel (KSTEPS x 8 fragments, 8KB/kstep) is staged in LDS once
// per block by a cooperative copy: global B traffic drops 8x (one copy per
// block instead of per wave) and each WMMA's operand dependency becomes a
// short ds_load (s_wait_dscnt) instead of a full global-load round trip.
// LDS fragment reads are 1024B contiguous per wave -> bank-conflict-free.
// Epilogue (decode_edge fusion): over M = B*N*N,
//   rowAdd indexed by (m>>6)  = b*N+i   (fa broadcast over cols)
//   colAdd indexed by (m>>12)*64+(m&63) = b*N+j (fb broadcast over rows)
// C/D fragment layout: element e -> row = 16*mt + (lane>>4)*8 + e,
//                               col = nt*16 + (lane&15).
// ---------------------------------------------------------------------------
template <int KSTEPS>
__global__ __launch_bounds__(256)
void gemm_n128_kernel(const v16h* __restrict__ Ap, const v16h* __restrict__ Bp,
                      const float* __restrict__ bias,
                      const float* __restrict__ rowAdd,
                      const float* __restrict__ colAdd,
                      float* __restrict__ C)
{
    extern __shared__ char smem[];                 // KSTEPS * 8192 bytes
    v16h* Bs = (v16h*)smem;

    // Cooperative B-panel copy: KSTEPS*512 uint4 (16B) chunks over 256 threads.
    {
        const uint4* gsrc = (const uint4*)Bp;
        uint4* sdst = (uint4*)smem;
#pragma unroll
        for (int it = 0; it < KSTEPS * 2; ++it) {
            int idx = it * 256 + threadIdx.x;
            sdst[idx] = gsrc[idx];
        }
    }
    __syncthreads();

    int wave = threadIdx.x >> 5;
    int lane = threadIdx.x & 31;
    int mt = blockIdx.x * 8 + wave;

    v8f zero = {0.f, 0.f, 0.f, 0.f, 0.f, 0.f, 0.f, 0.f};
    v8f acc[8];
#pragma unroll
    for (int nt = 0; nt < 8; ++nt) acc[nt] = zero;

    const v16h* Abase = Ap + (size_t)mt * KSTEPS * 32 + lane;
#pragma unroll
    for (int kt = 0; kt < KSTEPS; ++kt) {
        if (kt + 1 < KSTEPS)
            __builtin_prefetch((const void*)(Abase + (kt + 1) * 32), 0, 0);
        v16h a = Abase[kt * 32];
        const v16h* Bb = Bs + kt * 256 + lane;
#pragma unroll
        for (int nt = 0; nt < 8; ++nt) {
            v16h b = Bb[nt * 32];
            acc[nt] = __builtin_amdgcn_wmma_f32_16x16x32_f16(
                false, a, false, b, (short)0, acc[nt], false, false);
        }
    }

    int rbase = mt * 16 + (lane >> 4) * 8;
    int col0 = lane & 15;
#pragma unroll
    for (int nt = 0; nt < 8; ++nt) {
        int col = nt * 16 + col0;
        float badd = bias ? bias[col] : 0.f;
#pragma unroll
        for (int e = 0; e < 8; ++e) {
            int row = rbase + e;
            float v = acc[nt][e] + badd;
            if (rowAdd) v += rowAdd[(size_t)(row >> 6) * 128 + col];
            if (colAdd) v += colAdd[((size_t)(row >> 12) * 64 + (row & 63)) * 128 + col];
            C[(size_t)row * 128 + col] = v;
        }
    }
}

// ---------------------------------------------------------------------------
// scores_kernel: s[b,h,i,j] = 0.125*(q[b,i,h,:].(k[b,j,h,:]+ek[b,i,j,h,:]))
// masked (-1e9), softmax over j -> probs[b,h,i,j]. Block per (b,i), 128 thr,
// thread t: h=t>>6, j=t&63. Per-thread dot uses float4 (b128) loads.
// ---------------------------------------------------------------------------
__global__ __launch_bounds__(128)
void scores_kernel(const float* __restrict__ q, const float* __restrict__ k,
                   const float* __restrict__ ek, const float* __restrict__ mask,
                   int maskT, float* __restrict__ probs)
{
    int bi = blockIdx.x;
    int b = bi >> 6, i = bi & 63;
    int t = threadIdx.x;
    int h = t >> 6, j = t & 63;
    __shared__ __align__(16) float qs[128];
    __shared__ float sv[128];
    qs[t] = q[(size_t)bi * 128 + t];
    __syncthreads();

    const float4* k4 = (const float4*)(k + ((size_t)(b * 64 + j)) * 128 + h * 64);
    const float4* e4 = (const float4*)(ek + ((size_t)bi * 64 + j) * 128 + h * 64);
    const float4* q4 = (const float4*)(qs + h * 64);
    float acc = 0.f;
#pragma unroll
    for (int d = 0; d < 16; ++d) {
        float4 kv = k4[d], evv = e4[d], qv = q4[d];
        acc += qv.x * (kv.x + evv.x) + qv.y * (kv.y + evv.y)
             + qv.z * (kv.z + evv.z) + qv.w * (kv.w + evv.w);
    }

    float mv = maskT ? mask[((size_t)b * 64 + j) * 64 + i]
                     : mask[((size_t)b * 64 + i) * 64 + j];
    float s = (mv < 1.f) ? -1e9f : acc * 0.125f;   // scale = 1/sqrt(64)
    sv[t] = s;
    __syncthreads();

    float mx = -INFINITY;
    for (int jj = 0; jj < 64; ++jj) mx = fmaxf(mx, sv[h * 64 + jj]);
    float sum = 0.f;
    for (int jj = 0; jj < 64; ++jj) sum += expf(sv[h * 64 + jj] - mx);
    probs[(((size_t)b * 2 + h) * 64 + i) * 64 + j] = expf(s - mx) / sum;
}

// ---------------------------------------------------------------------------
// attn_out_kernel: o[b,i,h,d] = sum_j p[b,h,i,j]*(v[b,j,h,d]+ev[b,i,j,h,d])
// writes into o_cat[b*N+i, colOff + h*64+d]. Block per (b,i), thread t=h*64+d,
// ev/v reads fully coalesced (512B per j).
// ---------------------------------------------------------------------------
__global__ __launch_bounds__(128)
void attn_out_kernel(const float* __restrict__ probs, const float* __restrict__ v,
                     const float* __restrict__ ev, float* __restrict__ o_cat,
                     int colOff)
{
    int bi = blockIdx.x;
    int b = bi >> 6, i = bi & 63;
    int t = threadIdx.x;
    int h = t >> 6;
    __shared__ float p[128];
    p[t] = probs[(((size_t)b * 2 + h) * 64 + i) * 64 + (t & 63)];
    __syncthreads();

    const float* vb = v + (size_t)b * 64 * 128;
    const float* eb = ev + (size_t)bi * 64 * 128;
    float acc = 0.f;
#pragma unroll 4
    for (int j = 0; j < 64; ++j) {
        float pj = p[h * 64 + j];
        acc += pj * (vb[(size_t)j * 128 + t] + eb[(size_t)j * 128 + t]);
    }
    o_cat[(size_t)bi * 256 + colOff + t] = acc;
}

// ---------------------------------------------------------------------------
// dot_nodes_kernel: out[r] = f[r,:].w[off:off+128]  (qnet row/col terms)
// One wave per row, shuffle reduce.
// ---------------------------------------------------------------------------
__global__ __launch_bounds__(256)
void dot_nodes_kernel(const float* __restrict__ f, const float* __restrict__ w,
                      int off, float* __restrict__ out)
{
    int r = blockIdx.x * 8 + (threadIdx.x >> 5);
    int lane = threadIdx.x & 31;
    float acc = 0.f;
#pragma unroll
    for (int d = lane; d < 128; d += 32) acc += f[(size_t)r * 128 + d] * w[off + d];
#pragma unroll
    for (int s = 16; s > 0; s >>= 1) acc += __shfl_xor(acc, s, 32);
    if (lane == 0) out[r] = acc;
}

// ---------------------------------------------------------------------------
// qnet_edge_kernel: action[m] = ec[m,:].w[256:384] + rowdot[b*N+i]
//                              + coldot[b*N+j] + bias.  One wave per m.
// ---------------------------------------------------------------------------
__global__ __launch_bounds__(256)
void qnet_edge_kernel(const float* __restrict__ ec, const float* __restrict__ w,
                      const float* __restrict__ rowdot,
                      const float* __restrict__ coldot,
                      const float* __restrict__ bias, float* __restrict__ action)
{
    int m = blockIdx.x * 8 + (threadIdx.x >> 5);
    int lane = threadIdx.x & 31;
    float acc = 0.f;
#pragma unroll
    for (int d = lane; d < 128; d += 32) acc += ec[(size_t)m * 128 + d] * w[256 + d];
#pragma unroll
    for (int s = 16; s > 0; s >>= 1) acc += __shfl_xor(acc, s, 32);
    if (lane == 0)
        action[m] = acc + rowdot[m >> 6] + coldot[(m >> 12) * 64 + (m & 63)] + bias[0];
}

// ---------------------------------------------------------------------------
// masked_softmax_kernel: per batch b, softmax over 4096 masked logits.
// ---------------------------------------------------------------------------
__global__ __launch_bounds__(256)
void masked_softmax_kernel(const float* __restrict__ action,
                           const float* __restrict__ mask, float* __restrict__ out)
{
    int b = blockIdx.x, t = threadIdx.x;
    __shared__ float red[256];
    const float* a  = action + (size_t)b * 4096;
    const float* mk = mask + (size_t)b * 4096;

    float mx = -INFINITY;
    for (int idx = t; idx < 4096; idx += 256) {
        float v = (mk[idx] < 1.f) ? -1e9f : a[idx];
        mx = fmaxf(mx, v);
    }
    red[t] = mx; __syncthreads();
    for (int s = 128; s > 0; s >>= 1) { if (t < s) red[t] = fmaxf(red[t], red[t + s]); __syncthreads(); }
    mx = red[0]; __syncthreads();

    float sum = 0.f;
    for (int idx = t; idx < 4096; idx += 256) {
        float v = (mk[idx] < 1.f) ? -1e9f : a[idx];
        sum += expf(v - mx);
    }
    red[t] = sum; __syncthreads();
    for (int s = 128; s > 0; s >>= 1) { if (t < s) red[t] += red[t + s]; __syncthreads(); }
    sum = red[0]; __syncthreads();

    float inv = 1.f / sum;
    for (int idx = t; idx < 4096; idx += 256) {
        float v = (mk[idx] < 1.f) ? -1e9f : a[idx];
        out[(size_t)b * 4096 + idx] = expf(v - mx) * inv;
    }
}

// ===========================================================================
// Host orchestration.
// Input flattening assumption: top-level setup_inputs() insertion order;
// nested param dicts in jax tree (sorted-key) order:
//  0..7 : f1, f2, e1, e2, ec, m1, m2, mc
//  8..51: attn_params  na1_1,na1_2,na2_1,na2_2 (11 leaves each):
//         out.b, out.w(128x256), wek_c, wek_s, wev_c, wev_s,
//         wk_c, wk_s, wq, wv_c, wv_s   (all (128,128) unless noted)
// 52..65: lin_params coop1(b,w), coop2(b,w), qnet(b,w(1,384)),
//         task1_1..task2_2 (b, w(128,384))
// ===========================================================================
extern "C" void kernel_launch(void* const* d_in, const int* in_sizes, int n_in,
                              void* d_out, int out_size, void* d_ws, size_t ws_size,
                              hipStream_t stream)
{
    (void)in_sizes; (void)n_in; (void)out_size; (void)ws_size;

    const float* f1_in = (const float*)d_in[0];
    const float* f2_in = (const float*)d_in[1];
    const float* e1_in = (const float*)d_in[2];
    const float* e2_in = (const float*)d_in[3];
    const float* ec_in = (const float*)d_in[4];
    const float* m1 = (const float*)d_in[5];
    const float* m2 = (const float*)d_in[6];
    const float* mc = (const float*)d_in[7];

    const int OUT_B = 0, OUT_W = 1, WEK_C = 2, WEK_S = 3, WEV_C = 4, WEV_S = 5,
              WK_C = 6, WK_S = 7, WQ = 8, WV_C = 9, WV_S = 10;
    auto AP = [&](int blk, int leaf) { return (const float*)d_in[8 + blk * 11 + leaf]; };
    auto LPp = [&](int idx) { return (const float*)d_in[52 + idx]; };
    // lin leaves: 0 coop1.b 1 coop1.w 2 coop2.b 3 coop2.w 4 qnet.b 5 qnet.w
    //             6 t11.b 7 t11.w 8 t12.b 9 t12.w 10 t21.b 11 t21.w 12 t22.b 13 t22.w

    // ---- workspace bump allocator (total ~320 MB) ----
    char* wp = (char*)d_ws;
    auto alloc = [&](size_t bytes) -> void* {
        void* r = (void*)wp; wp += (bytes + 255) & ~(size_t)255; return r;
    };
    const size_t EDGE_F32 = (size_t)M_EDGE * 128 * 4;
    float* ec   = (float*)alloc(EDGE_F32);
    float* e1   = (float*)alloc(EDGE_F32);
    float* e2   = (float*)alloc(EDGE_F32);
    float* proj = (float*)alloc(EDGE_F32);                     // EK or EV (f32)
    _Float16* packAbig = (_Float16*)alloc((size_t)M_EDGE * 128 * 2);
    float* probs_s = (float*)alloc((size_t)BB * 2 * NN * NN * 4);
    float* probs_c = (float*)alloc((size_t)BB * 2 * NN * NN * 4);
    const size_t NODE_F32 = (size_t)M_NODE * 128 * 4;
    float* qbuf = (float*)alloc(NODE_F32);
    float* kbuf = (float*)alloc(NODE_F32);
    float* vbuf = (float*)alloc(NODE_F32);
    float* rowterm = (float*)alloc(NODE_F32);
    float* colterm = (float*)alloc(NODE_F32);
    float* f1_1 = (float*)alloc(NODE_F32);
    float* f2_1 = (float*)alloc(NODE_F32);
    float* f1_2 = (float*)alloc(NODE_F32);
    float* f2_2 = (float*)alloc(NODE_F32);
    float* o_cat = (float*)alloc((size_t)M_NODE * 256 * 4);
    _Float16* npA1 = (_Float16*)alloc((size_t)M_NODE * 128 * 2);
    _Float16* npA2 = (_Float16*)alloc((size_t)M_NODE * 128 * 2);
    _Float16* npQ  = (_Float16*)alloc((size_t)M_NODE * 128 * 2);
    _Float16* npKV = (_Float16*)alloc((size_t)M_NODE * 128 * 2);
    _Float16* npO  = (_Float16*)alloc((size_t)M_NODE * 256 * 2);
    _Float16* Bslot = (_Float16*)alloc((size_t)256 * 128 * 2);
    float* action = (float*)alloc((size_t)M_EDGE * 4);
    float* rowdot = (float*)alloc((size_t)M_NODE * 4);
    float* coldot = (float*)alloc((size_t)M_NODE * 4);

    // ---- launch helpers (stream-ordered; Bslot reuse is safe) ----
    auto packA = [&](const float* src, _Float16* dst, int M, int K, int tr) {
        int frags = (M / 16) * (K / 32);
        pack_a_kernel<<<dim3((frags + 7) / 8), dim3(256), 0, stream>>>(src, dst, M, K, tr);
    };
    auto packB = [&](const float* w, int K, int ldw, int off, int mode) {
        pack_b_kernel<<<dim3(K / 32), dim3(256), 0, stream>>>(w, Bslot, K, ldw, off, mode);
    };
    auto gemm = [&](const _Float16* A, const float* bias, const float* rAdd,
                    const float* cAdd, float* C, int M, int K) {
        if (K == 128)
            gemm_n128_kernel<4><<<dim3(M / 128), dim3(256), 4 * 8192, stream>>>(
                (const v16h*)A, (const v16h*)Bslot, bias, rAdd, cAdd, C);
        else
            gemm_n128_kernel<8><<<dim3(M / 128), dim3(256), 8 * 8192, stream>>>(
                (const v16h*)A, (const v16h*)Bslot, bias, rAdd, cAdd, C);
    };

    // decode_edge: out = fa@Wa^T (row bcast) + fb@Wb^T (col bcast) + edge@We^T + b
    auto decode = [&](const float* edge_src, const _Float16* faP, const _Float16* fbP,
                      const float* w, const float* b, float* out) {
        packB(w, 128, 384, 0, 1);   gemm(faP, nullptr, nullptr, nullptr, rowterm, M_NODE, 128);
        packB(w, 128, 384, 128, 1); gemm(fbP, nullptr, nullptr, nullptr, colterm, M_NODE, 128);
        packA(edge_src, packAbig, M_EDGE, 128, 0);
        packB(w, 128, 384, 256, 1);
        gemm(packAbig, b, rowterm, colterm, out, M_EDGE, 128);
    };

    auto branch = [&](const _Float16* fP, const float* edge_f32, int eTr,
                      const float* mask, int mTr, const float* wk, const float* wv,
                      const float* wek, const float* wev, float* probs, int colOff) {
        packB(wk, 128, 128, 0, 0);  gemm(fP, nullptr, nullptr, nullptr, kbuf, M_NODE, 128);
        packB(wv, 128, 128, 0, 0);  gemm(fP, nullptr, nullptr, nullptr, vbuf, M_NODE, 128);
        packA(edge_f32, packAbig, M_EDGE, 128, eTr);
        packB(wek, 128, 128, 0, 0); gemm(packAbig, nullptr, nullptr, nullptr, proj, M_EDGE, 128);
        scores_kernel<<<dim3(M_NODE), dim3(128), 0, stream>>>(qbuf, kbuf, proj, mask, mTr, probs);
        packB(wev, 128, 128, 0, 0); gemm(packAbig, nullptr, nullptr, nullptr, proj, M_EDGE, 128);
        attn_out_kernel<<<dim3(M_NODE), dim3(128), 0, stream>>>(probs, vbuf, proj, o_cat, colOff);
    };

    auto edge_attn = [&](const float* fq, const float* fkv, const float* e_self,
                         int ecTr, const float* m_self, int mcTr, int blk, float* fout) {
        packA(fq, npQ, M_NODE, 128, 0);
        packA(fkv, npKV, M_NODE, 128, 0);
        packB(AP(blk, WQ), 128, 128, 0, 0);
        gemm(npQ, nullptr, nullptr, nullptr, qbuf, M_NODE, 128);
        branch(npQ, e_self, 0, m_self, 0, AP(blk, WK_S), AP(blk, WV_S),
               AP(blk, WEK_S), AP(blk, WEV_S), probs_s, 0);
        branch(npKV, ec, ecTr, mc, mcTr, AP(blk, WK_C), AP(blk, WV_C),
               AP(blk, WEK_C), AP(blk, WEV_C), probs_c, 128);
        packA(o_cat, npO, M_NODE, 256, 0);
        packB(AP(blk, OUT_W), 256, 256, 0, 1);
        gemm(npO, AP(blk, OUT_B), nullptr, nullptr, fout, M_NODE, 256);
    };

    // ---------------- forward pass ----------------
    // Layer-1 decode
    packA(f1_in, npA1, M_NODE, 128, 0);
    packA(f2_in, npA2, M_NODE, 128, 0);
    decode(ec_in, npA1, npA2, LPp(1), LPp(0), ec);    // coop1
    decode(e1_in, npA1, npA1, LPp(7), LPp(6), e1);    // task1_1
    decode(e2_in, npA2, npA2, LPp(9), LPp(8), e2);    // task1_2
    // Layer-1 attention
    edge_attn(f1_in, f2_in, e1, 0, m1, 0, 0, f1_1);   // na1_1
    edge_attn(f2_in, f1_in, e2, 1, m2, 1, 1, f2_1);   // na1_2 (ec^T, mc^T)
    // Layer-2 decode (in place; A is consumed via packed f16 copy first)
    packA(f1_1, npA1, M_NODE, 128, 0);
    packA(f2_1, npA2, M_NODE, 128, 0);
    decode(ec, npA1, npA2, LPp(3), LPp(2), ec);       // coop2
    decode(e1, npA1, npA1, LPp(11), LPp(10), e1);     // task2_1
    decode(e2, npA2, npA2, LPp(13), LPp(12), e2);     // task2_2
    // Layer-2 attention
    edge_attn(f1_1, f2_1, e1, 0, m1, 0, 2, f1_2);     // na2_1
    edge_attn(f2_1, f1_1, e2, 1, m2, 1, 3, f2_2);     // na2_2
    // q-net head + masked softmax
    dot_nodes_kernel<<<dim3(M_NODE / 8), dim3(256), 0, stream>>>(f1_2, LPp(5), 0, rowdot);
    dot_nodes_kernel<<<dim3(M_NODE / 8), dim3(256), 0, stream>>>(f2_2, LPp(5), 128, coldot);
    qnet_edge_kernel<<<dim3(M_EDGE / 8), dim3(256), 0, stream>>>(ec, LPp(5), rowdot, coldot, LPp(4), action);
    masked_softmax_kernel<<<dim3(BB), dim3(256), 0, stream>>>(action, mc, (float*)d_out);
}
